// series_decomp_39324720562741
// MI455X (gfx1250) — matrix operand order repaired and checked by reading, not compiled, not run on hardware
//
#include <hip/hip_runtime.h>
#include <hip/hip_bf16.h>
#include <math.h>

typedef __attribute__((ext_vector_type(16))) _Float16 v16h;
typedef __attribute__((ext_vector_type(8)))  float    v8f;

#define CDIM  256
#define TLEN  2048
#define BATCH 8
#define HEADS 4
#define HDN   64

#define EPI_NONE  0
#define EPI_TANH  1
#define EPI_SIG   2
#define EPI_RELU2 3
#define EPI_WDEC  4

// ---------------- LayerNorm over C=256 (one block per row) ----------------
__global__ void k_layernorm(const float* __restrict__ X, const float* __restrict__ w,
                            const float* __restrict__ b, float* __restrict__ O) {
  __shared__ float s1[CDIM], s2[CDIM];
  int row = blockIdx.x, c = threadIdx.x;
  float v = X[(size_t)row * CDIM + c];
  s1[c] = v; s2[c] = v * v;
  __syncthreads();
  for (int st = CDIM / 2; st >= 1; st >>= 1) {
    if (c < st) { s1[c] += s1[c + st]; s2[c] += s2[c + st]; }
    __syncthreads();
  }
  float mu  = s1[0] * (1.f / CDIM);
  float var = s2[0] * (1.f / CDIM) - mu * mu;
  O[(size_t)row * CDIM + c] = (v - mu) * rsqrtf(var + 1e-5f) * w[c] + b[c];
}

// ---------------- fused epilogues ----------------
__device__ __forceinline__ float epi_apply(float acc, int epi,
                                           const float* bias, int n) {
  switch (epi) {
    case EPI_TANH:  return tanhf(acc);
    case EPI_SIG: { float s = (bias ? bias[n] : 0.f) + acc;
                    return 1.f / (1.f + expf(-s)); }
    case EPI_RELU2: { float r = fmaxf(acc, 0.f); return r * r; }
    case EPI_WDEC: { float s = (bias ? bias[n] : 0.f) + acc;
                     float w = -log1pf(expf(-s)) - 0.5f;   // -softplus(-s)-0.5
                     return expf(-expf(w)); }              // decay
    default: return acc;
  }
}

// ---------------- WMMA GEMM: Y[M,N] = mix(X)[M,K] @ W[K,N] ----------------
// mix (if coef!=null): per row m=(b,t): x + (x[t-1]-x[t])*coef[c], zero at t=0.
// Tile 64x64x32, 4 waves / 128 threads; f16 operands, f32 accumulate.
__global__ void __launch_bounds__(128)
k_gemm_wmma(const float* __restrict__ X, const float* __restrict__ coef,
            const float* __restrict__ W, float* __restrict__ Y,
            int M, int K, int N, int T, int epi, const float* __restrict__ bias) {
  __shared__ _Float16 As[64][34];   // [m][k]
  __shared__ _Float16 Bs[64][34];   // [n][k] (transposed stage)
  int tid  = threadIdx.x;
  int lane = tid & 31, wv = tid >> 5, hi = lane >> 4, l15 = lane & 15;
  int m0 = blockIdx.y * 64, n0 = blockIdx.x * 64;
  v8f acc[4] = {};

  for (int k0 = 0; k0 < K; k0 += 32) {
    if (k0 + 32 < K)   // gfx1250 global_prefetch_b8 on the next W tile
      __builtin_prefetch(&W[(size_t)(k0 + 32) * N + n0], 0, 1);
    // A tile: 64x32, with fused token-shift mix
    #pragma unroll
    for (int i = 0; i < 16; ++i) {
      int e = tid + i * 128;
      int r = e >> 5, cc = e & 31;
      int m = m0 + r, kk = k0 + cc;
      float v = X[(size_t)m * K + kk];
      if (coef) {
        int t = m % T;
        float prev = (t > 0) ? X[(size_t)(m - 1) * K + kk] : 0.f;
        v = v + (prev - v) * coef[kk];
      }
      As[r][cc] = (_Float16)v;
    }
    // B tile: W[k0+kr][n0+nc] -> Bs[nc][kr]
    #pragma unroll
    for (int i = 0; i < 16; ++i) {
      int e = tid + i * 128;
      int kr = e >> 6, nc = e & 63;
      Bs[nc][kr] = (_Float16)W[(size_t)(k0 + kr) * N + (n0 + nc)];
    }
    __syncthreads();
    // A fragment per ISA 16-bit A 16x32 layout
    v16h af;
    #pragma unroll
    for (int j = 0; j < 16; ++j) {
      int vg = j >> 1, wb = j & 1;
      int kl = ((vg < 4) ? 0 : 16) + hi * 8 + (vg & 3) * 2 + wb;
      af[j] = As[wv * 16 + l15][kl];
    }
    #pragma unroll
    for (int g = 0; g < 4; ++g) {
      v16h bf;   // B: lane=l15 -> N, K = (j<8 ? 0 : 8) + hi*8 + j
      #pragma unroll
      for (int j = 0; j < 16; ++j)
        bf[j] = Bs[g * 16 + l15][((j < 8) ? 0 : 8) + hi * 8 + j];
      acc[g] = __builtin_amdgcn_wmma_f32_16x16x32_f16(
          false, af, false, bf, (short)0, acc[g], false, false);
    }
    __syncthreads();
  }
  // C layout: VGPR rr -> M = rr + hi*8, N = l15
  #pragma unroll
  for (int g = 0; g < 4; ++g)
    #pragma unroll
    for (int rr = 0; rr < 8; ++rr) {
      int m = m0 + wv * 16 + hi * 8 + rr;
      int n = n0 + g * 16 + l15;
      Y[(size_t)m * N + n] = epi_apply(acc[g][rr], epi, bias, n);
    }
}

// ---------------- kk normalize + k gating (one block per (b,t)) -----------
__global__ void __launch_bounds__(256)
k_prep(float* __restrict__ k, const float* __restrict__ aa,
       const float* __restrict__ k_k, const float* __restrict__ k_a,
       float* __restrict__ kk_out) {
  __shared__ float red[256];
  int row = blockIdx.x, c = threadIdx.x;
  size_t idx = (size_t)row * CDIM + c;
  float kv = k[idx], av = aa[idx];
  float kkv = kv * k_k[c];
  red[c] = kkv * kkv;
  __syncthreads();
  for (int s = 32; s >= 1; s >>= 1) {
    if ((c & 63) < s) red[c] += red[c + s];
    __syncthreads();
  }
  float nrm = sqrtf(red[c & ~63]);
  kk_out[idx] = kkv / fmaxf(nrm, 1e-12f);
  k[idx] = kv * (1.f + (av - 1.f) * k_a[c]);
}

// ---------------- sequential RWKV7 scan: one block per (b,h) --------------
// thread (i = tid>>2, jq = tid&3) owns 16 state cols of row i in registers.
__global__ void __launch_bounds__(256)
k_scan(const float* __restrict__ r, const float* __restrict__ wd,
       const float* __restrict__ k, const float* __restrict__ v,
       const float* __restrict__ kk, const float* __restrict__ aa,
       float* __restrict__ y) {
  int b = blockIdx.x / HEADS, h = blockIdx.x % HEADS;
  int tid = threadIdx.x;
  int i = tid >> 2, jq = tid & 3, j0 = jq * 16;
  __shared__ float vr[64], vw[64], vk[64], vv[64], va[64], vb[64];
  float st[16];
  #pragma unroll
  for (int q = 0; q < 16; ++q) st[q] = 0.f;

  for (int t = 0; t < TLEN; ++t) {
    size_t base = ((size_t)(b * TLEN + t)) * CDIM + h * HDN;
    if (tid < 64) {
      vr[tid] = r[base + tid]; vw[tid] = wd[base + tid];
      vk[tid] = k[base + tid]; vv[tid] = v[base + tid];
      float kkv = kk[base + tid];
      va[tid] = -kkv; vb[tid] = kkv * aa[base + tid];
    }
    __syncthreads();
    float sa = 0.f;
    #pragma unroll
    for (int q = 0; q < 16; ++q) sa += st[q] * va[j0 + q];
    sa += __shfl_xor(sa, 1);
    sa += __shfl_xor(sa, 2);
    float vi = vv[i];
    float yp = 0.f;
    #pragma unroll
    for (int q = 0; q < 16; ++q) {
      int j = j0 + q;
      st[q] = st[q] * vw[j] + sa * vb[j] + vi * vk[j];
      yp += vr[j] * st[q];
    }
    yp += __shfl_xor(yp, 1);
    yp += __shfl_xor(yp, 2);
    if (jq == 0) y[base + i] = yp;
    __syncthreads();
  }
}

// ---------------- per-head GroupNorm + bonus + gate -----------------------
__global__ void __launch_bounds__(256)
k_post(const float* __restrict__ y, const float* __restrict__ r,
       const float* __restrict__ k, const float* __restrict__ v,
       const float* __restrict__ g, const float* __restrict__ r_k,
       const float* __restrict__ lnw, const float* __restrict__ lnb,
       float* __restrict__ attin) {
  __shared__ float s1[256], s2[256], s3[256];
  int row = blockIdx.x, c = threadIdx.x;
  size_t idx = (size_t)row * CDIM + c;
  float yv = y[idx], rv = r[idx], kv = k[idx];
  s1[c] = yv; s2[c] = yv * yv; s3[c] = rv * kv * r_k[c];
  __syncthreads();
  for (int s = 32; s >= 1; s >>= 1) {
    if ((c & 63) < s) { s1[c] += s1[c + s]; s2[c] += s2[c + s]; s3[c] += s3[c + s]; }
    __syncthreads();
  }
  int hb = c & ~63;
  float mu  = s1[hb] * (1.f / HDN);
  float var = s2[hb] * (1.f / HDN) - mu * mu;
  float yn  = (yv - mu) * rsqrtf(var + 64e-5f) * lnw[c] + lnb[c];
  attin[idx] = (yn + s3[hb] * v[idx]) * g[idx];
}

__global__ void k_add(const float* __restrict__ a, const float* __restrict__ b,
                      float* __restrict__ o, int n) {
  int i = blockIdx.x * blockDim.x + threadIdx.x;
  if (i < n) o[i] = a[i] + b[i];
}

// ---------------- final: inline moving average + combine ------------------
__global__ void k_final(const float* __restrict__ x, const float* __restrict__ x1,
                        const float* __restrict__ cm, const float* __restrict__ a_ptr,
                        float* __restrict__ out, int n) {
  int i = blockIdx.x * blockDim.x + threadIdx.x;
  if (i >= n) return;
  int c = i & (CDIM - 1);
  int m = i >> 8;
  int b = m / TLEN, t = m % TLEN;
  float s = 0.f;
  #pragma unroll
  for (int d = -12; d <= 12; ++d) {
    int tt = t + d; tt = tt < 0 ? 0 : (tt > TLEN - 1 ? TLEN - 1 : tt);
    s += x[((size_t)(b * TLEN + tt)) * CDIM + c];
  }
  float ma = s * (1.f / 25.f);
  float av = a_ptr[0];
  float lx = x1[i] + cm[i];
  float mm = ma * av + lx * (1.f - av);
  out[i] = x[i] - mm;                 // res
  out[(size_t)n + i] = mm;            // moving_mean
}

extern "C" void kernel_launch(void* const* d_in, const int* in_sizes, int n_in,
                              void* d_out, int out_size, void* d_ws, size_t ws_size,
                              hipStream_t stream) {
  (void)in_sizes; (void)n_in; (void)out_size; (void)ws_size;
  const float* x    = (const float*)d_in[0];
  const float* ln1w = (const float*)d_in[1];
  const float* ln1b = (const float*)d_in[2];
  const float* ln2w = (const float*)d_in[3];
  const float* ln2b = (const float*)d_in[4];
  const float* x_r  = (const float*)d_in[5];
  const float* x_w  = (const float*)d_in[6];
  const float* x_k  = (const float*)d_in[7];
  const float* x_v  = (const float*)d_in[8];
  const float* x_a  = (const float*)d_in[9];
  const float* x_g  = (const float*)d_in[10];
  const float* w0   = (const float*)d_in[11];
  const float* w1   = (const float*)d_in[12];
  const float* w2   = (const float*)d_in[13];
  const float* a0   = (const float*)d_in[14];
  const float* a1   = (const float*)d_in[15];
  const float* a2   = (const float*)d_in[16];
  const float* g1   = (const float*)d_in[17];
  const float* g2   = (const float*)d_in[18];
  const float* k_k  = (const float*)d_in[19];
  const float* k_a  = (const float*)d_in[20];
  const float* r_k  = (const float*)d_in[21];
  const float* Wr   = (const float*)d_in[22];
  const float* Wk   = (const float*)d_in[23];
  const float* Wv   = (const float*)d_in[24];
  const float* Wo   = (const float*)d_in[25];
  const float* lnxw = (const float*)d_in[26];
  const float* lnxb = (const float*)d_in[27];
  const float* cx_k = (const float*)d_in[28];
  const float* cWk  = (const float*)d_in[29];
  const float* cWv  = (const float*)d_in[30];
  const float* a_sc = (const float*)d_in[31];

  const int M = BATCH * TLEN;            // 16384
  const size_t MC = (size_t)M * CDIM;    // 4,194,304 floats
  float* ws    = (float*)d_ws;
  float* xln   = ws;
  float* rB    = ws + 1 * MC;
  float* kB    = ws + 2 * MC;
  float* vB    = ws + 3 * MC;
  float* wdec  = ws + 4 * MC;
  float* aaB   = ws + 5 * MC;
  float* gBuf  = ws + 6 * MC;
  float* kkB   = ws + 7 * MC;
  float* hid   = ws + 8 * MC;            // M*128 max
  float* yB    = ws + 8 * MC + MC / 2;   // later reused as x1
  float* attin = ws + 9 * MC + MC / 2;
  float* attB  = ws + 10 * MC + MC / 2;
  float* x2ln  = ws + 11 * MC + MC / 2;
  float* chid  = rB;                     // M*1024, overlaps dead r..wdec
  float* cout  = xln;                    // overlaps dead xln
  float* x1    = yB;                     // overlaps dead y

  auto gemm = [&](const float* X, const float* coef, const float* W, float* Y,
                  int K, int N, int epi, const float* bias) {
    dim3 grid(N / 64, M / 64);
    k_gemm_wmma<<<grid, 128, 0, stream>>>(X, coef, W, Y, M, K, N, TLEN, epi, bias);
  };

  k_layernorm<<<M, CDIM, 0, stream>>>(x, ln1w, ln1b, xln);

  gemm(xln, x_r, Wr, rB,   256, 256,  EPI_NONE,  nullptr);
  gemm(xln, x_k, Wk, kB,   256, 256,  EPI_NONE,  nullptr);
  gemm(xln, x_v, Wv, vB,   256, 256,  EPI_NONE,  nullptr);
  gemm(xln, x_w, w1, hid,  256, 64,   EPI_TANH,  nullptr);
  gemm(hid, nullptr, w2, wdec, 64, 256, EPI_WDEC, w0);
  gemm(xln, x_a, a1, hid,  256, 64,   EPI_NONE,  nullptr);
  gemm(hid, nullptr, a2, aaB, 64, 256, EPI_SIG,  a0);
  gemm(xln, x_g, g1, hid,  256, 128,  EPI_SIG,   nullptr);
  gemm(hid, nullptr, g2, gBuf, 128, 256, EPI_NONE, nullptr);

  k_prep<<<M, 256, 0, stream>>>(kB, aaB, k_k, k_a, kkB);
  k_scan<<<BATCH * HEADS, 256, 0, stream>>>(rB, wdec, kB, vB, kkB, aaB, yB);
  k_post<<<M, 256, 0, stream>>>(yB, rB, kB, vB, gBuf, r_k, lnxw, lnxb, attin);

  gemm(attin, nullptr, Wo, attB, 256, 256, EPI_NONE, nullptr);
  k_add<<<(int)((MC + 255) / 256), 256, 0, stream>>>(x, attB, x1, (int)MC);

  k_layernorm<<<M, CDIM, 0, stream>>>(x1, ln2w, ln2b, x2ln);
  gemm(x2ln, cx_k, cWk, chid, 256, 1024, EPI_RELU2, nullptr);
  gemm(chid, nullptr, cWv, cout, 1024, 256, EPI_NONE, nullptr);

  k_final<<<(int)((MC + 255) / 256), 256, 0, stream>>>(x, x1, cout, a_sc,
                                                       (float*)d_out, (int)MC);
}
